// GCNNet_72859825209830
// MI455X (gfx1250) — matrix operand order repaired and checked by reading, not compiled, not run on hardware
//
#include <hip/hip_runtime.h>
#include <cmath>

typedef __attribute__((ext_vector_type(16))) __bf16 v16bf;
typedef __attribute__((ext_vector_type(8)))  float  v8f;

static constexpr float GN_EPS = 1e-5f;
#define TPB 256

// ---------------------------------------------------------------------------
// Helpers
// ---------------------------------------------------------------------------
__device__ __forceinline__ int lower_bound_i(const int* a, int n, int key) {
  int lo = 0, hi = n;
  while (lo < hi) { int mid = (lo + hi) >> 1; if (a[mid] < key) lo = mid + 1; else hi = mid; }
  return lo;
}

// ---------------------------------------------------------------------------
// LDS-staged WMMA GEMM: C[M,N] = A[M,K] @ W[K,N] + bias (fp32 in/out, bf16 MACs)
//
// block = 256 threads = 8 waves; block tile 64(M) x 64(N); K stepped by 32.
// Wave w: wm = w&3 -> 16-row slab; wn = w>>2 -> 32-col slab = two 16x16 WMMAs
// sharing one A fragment. LDS tiles are K-contiguous so each fragment VGPR
// (a K-pair per the CDNA5 16-bit layouts) is a single aligned ds_load_b32.
// Staging is branchless (clamped index + select 0), coalesced 32B/thread.
// M must be a multiple of 64 (true for 49152 / 65536 / 1024). N may be ragged.
// ---------------------------------------------------------------------------
#define BM 64
#define BN 64
#define BK 32
#define LDK (BK + 2)   // +2 halfwords: break 16-bank row stride, keep 4B pair alignment

__global__ __launch_bounds__(256) void k_gemm_wmma(
    const float* __restrict__ A, const float* __restrict__ W,
    const float* __restrict__ bias, float* __restrict__ C,
    int M, int K, int N)
{
  __shared__ __bf16 As[BM][LDK];   // [m][k]
  __shared__ __bf16 Bs[BN][LDK];   // [n][k] (transposed so K-pairs are contiguous)

  const int tid  = threadIdx.x;
  const int lane = tid & 31;
  const int wave = tid >> 5;          // 0..7
  const int wm   = wave & 3;          // M sub-tile
  const int wn   = wave >> 2;         // N sub-tile (0/1)
  const int m0   = blockIdx.y * BM;
  const int n0   = blockIdx.x * BN;
  const int l15  = lane & 15;
  const int hi   = lane >> 4;         // lane half

  v8f acc0 = {};
  v8f acc1 = {};

  // staging coordinates (fixed per thread)
  const int ar = tid >> 2;            // A: row 0..63
  const int ac = (tid & 3) * 8;       // A: 8 consecutive K per thread
  const int bn = tid & 63;            // B: col 0..63
  const int bk = tid >> 6;            // B: k phase 0..3
  const int bcol  = n0 + bn;
  const int bci   = (bcol < N) ? bcol : 0;
  const bool bcin = (bcol < N);

  const int nsteps = (K + BK - 1) / BK;
  for (int step = 0; step < nsteps; ++step) {
    const int k0 = step * BK;

    // ---- stage A tile (64x32): branchless, coalesced 8 floats/thread
    {
      const float* Arow = A + (size_t)(m0 + ar) * K;
#pragma unroll
      for (int j = 0; j < 8; ++j) {
        const int kk = k0 + ac + j;
        const int ki = (kk < K) ? kk : 0;
        const float x = Arow[ki];
        As[ar][ac + j] = (__bf16)((kk < K) ? x : 0.0f);
      }
    }
    // ---- stage B tile: W[k0..k0+31][n0..n0+63] -> Bs[n][k]
    {
#pragma unroll
      for (int it = 0; it < 8; ++it) {
        const int kk = k0 + bk + it * 4;
        const int ki = (kk < K) ? kk : 0;
        const float x = W[(size_t)ki * N + bci];
        Bs[bn][bk + it * 4] = (__bf16)(((kk < K) && bcin) ? x : 0.0f);
      }
    }
    __syncthreads();

    // ---- fragments from LDS (guard-free; one ds_load_b32 per K-pair)
    union { unsigned u[8]; v16bf v; } fa, fb0, fb1;
    const int am = wm * 16 + l15;
#pragma unroll
    for (int v = 0; v < 8; ++v) {
      const int k = (v < 4) ? (hi * 8 + 2 * v) : (16 + hi * 8 + 2 * (v - 4));
      fa.u[v] = *(const unsigned*)&As[am][k];
    }
    const int nb0 = wn * 32 + l15;
#pragma unroll
    for (int v = 0; v < 8; ++v) {
      const int k = (v < 4) ? (hi * 8 + 2 * v) : (16 + hi * 8 + 2 * (v - 4));
      fb0.u[v] = *(const unsigned*)&Bs[nb0][k];
      fb1.u[v] = *(const unsigned*)&Bs[nb0 + 16][k];
    }

    acc0 = __builtin_amdgcn_wmma_f32_16x16x32_bf16(false, fa.v, false, fb0.v,
                                                   (short)0, acc0, false, false);
    acc1 = __builtin_amdgcn_wmma_f32_16x16x32_bf16(false, fa.v, false, fb1.v,
                                                   (short)0, acc1, false, false);
    __syncthreads();
  }

  // ---- store (C/D layout: VGPR r -> row +r+hi*8, col = lane&15)
  const int row0 = m0 + wm * 16 + hi * 8;
  const int col0 = n0 + wn * 32 + l15;
  const int col1 = col0 + 16;
  if (col0 < N) {
    const float bv = bias ? bias[col0] : 0.0f;
#pragma unroll
    for (int r = 0; r < 8; ++r) C[(size_t)(row0 + r) * N + col0] = acc0[r] + bv;
  }
  if (col1 < N) {
    const float bv = bias ? bias[col1] : 0.0f;
#pragma unroll
    for (int r = 0; r < 8; ++r) C[(size_t)(row0 + r) * N + col1] = acc1[r] + bv;
  }
}

// ---------------------------------------------------------------------------
// Graph-structure kernels
// ---------------------------------------------------------------------------
__global__ void k_zero_f32(float* __restrict__ p, long long n) {
  long long i = (long long)blockIdx.x * blockDim.x + threadIdx.x;
  if (i < n) p[i] = 0.0f;
}

__global__ void k_degree(const int* __restrict__ dst, float* __restrict__ deg, int E) {
  int i = blockIdx.x * blockDim.x + threadIdx.x;
  if (i < E) unsafeAtomicAdd(&deg[dst[i]], 1.0f);
}

__global__ void k_dinv(float* __restrict__ deg, int N) {
  int i = blockIdx.x * blockDim.x + threadIdx.x;
  if (i < N) { float d = deg[i]; deg[i] = (d > 0.0f) ? rsqrtf(d) : 0.0f; }
}

__global__ void k_edge_norm(const int* __restrict__ src, const int* __restrict__ dst,
                            const float* __restrict__ dinv, float* __restrict__ nrm, int E) {
  int i = blockIdx.x * blockDim.x + threadIdx.x;
  if (i < E) nrm[i] = dinv[src[i]] * dinv[dst[i]];
}

__global__ void k_fill_bias(float* __restrict__ G, const float* __restrict__ bias,
                            long long total, int F) {
  long long i = (long long)blockIdx.x * blockDim.x + threadIdx.x;
  if (i < total) G[i] = bias[(int)(i % F)];
}

// out[dst] += H[src] * norm   (one thread per (edge, feature))
__global__ void k_scatter_add(const float* __restrict__ H, const int* __restrict__ src,
                              const int* __restrict__ dst, const float* __restrict__ nrm,
                              float* __restrict__ G, long long total, int F) {
  long long i = (long long)blockIdx.x * blockDim.x + threadIdx.x;
  if (i >= total) return;
  const int e = (int)(i / F);
  const int f = (int)(i % F);
  const float v = H[(size_t)src[e] * F + f] * nrm[e];
  unsafeAtomicAdd(&G[(size_t)dst[e] * F + f], v);
}

// ---------------------------------------------------------------------------
// GraphNorm (+ReLU). One block per graph; batch[] is sorted -> binary search.
// ---------------------------------------------------------------------------
__global__ void k_graphnorm_relu(const float* __restrict__ X, float* __restrict__ Y,
                                 const int* __restrict__ batch, int Ntot, int F,
                                 const float* __restrict__ gamma, const float* __restrict__ beta,
                                 const float* __restrict__ alpha) {
  __shared__ float smean[512];
  __shared__ float svar[512];
  __shared__ int srange[2];
  const int g = blockIdx.x;
  if (threadIdx.x == 0) {
    srange[0] = lower_bound_i(batch, Ntot, g);
    srange[1] = lower_bound_i(batch, Ntot, g + 1);
  }
  __syncthreads();
  const int s = srange[0], e = srange[1];
  const float cnt = (float)max(e - s, 1);

  for (int f = threadIdx.x; f < F; f += blockDim.x) {
    float acc = 0.0f;
    for (int n = s; n < e; ++n) acc += X[(size_t)n * F + f];
    smean[f] = acc / cnt;
  }
  __syncthreads();
  for (int f = threadIdx.x; f < F; f += blockDim.x) {
    const float m = alpha[f] * smean[f];
    float acc = 0.0f;
    for (int n = s; n < e; ++n) { float d = X[(size_t)n * F + f] - m; acc += d * d; }
    svar[f] = acc / cnt;
  }
  __syncthreads();
  const int total = (e - s) * F;
  for (int t = threadIdx.x; t < total; t += blockDim.x) {
    const int n = s + t / F;
    const int f = t % F;
    const float xc = X[(size_t)n * F + f] - alpha[f] * smean[f];
    float y = gamma[f] * xc * rsqrtf(svar[f] + GN_EPS) + beta[f];
    Y[(size_t)n * F + f] = fmaxf(y, 0.0f);
  }
}

// mean pool: one block per graph
__global__ void k_mean_pool(const float* __restrict__ X, float* __restrict__ P,
                            const int* __restrict__ batch, int Ntot, int F) {
  __shared__ int srange[2];
  const int g = blockIdx.x;
  if (threadIdx.x == 0) {
    srange[0] = lower_bound_i(batch, Ntot, g);
    srange[1] = lower_bound_i(batch, Ntot, g + 1);
  }
  __syncthreads();
  const int s = srange[0], e = srange[1];
  const float inv = 1.0f / (float)max(e - s, 1);
  for (int f = threadIdx.x; f < F; f += blockDim.x) {
    float acc = 0.0f;
    for (int n = s; n < e; ++n) acc += X[(size_t)n * F + f];
    P[(size_t)g * F + f] = acc * inv;
  }
}

// BatchNorm over M rows (+ReLU). One thread per feature column.
__global__ void k_bn_relu(const float* __restrict__ X, float* __restrict__ Y,
                          const float* __restrict__ gamma, const float* __restrict__ beta,
                          int M, int F) {
  const int f = blockIdx.x * blockDim.x + threadIdx.x;
  if (f >= F) return;
  float s = 0.0f, s2 = 0.0f;
  for (int r = 0; r < M; ++r) { float x = X[(size_t)r * F + f]; s += x; s2 += x * x; }
  const float m = s / (float)M;
  const float var = s2 / (float)M - m * m;
  const float inv = rsqrtf(var + GN_EPS);
  const float ga = gamma[f], be = beta[f];
  for (int r = 0; r < M; ++r) {
    float x = X[(size_t)r * F + f];
    Y[(size_t)r * F + f] = fmaxf(ga * (x - m) * inv + be, 0.0f);
  }
}

__global__ void k_concat(const float* __restrict__ Xd, const float* __restrict__ Xt,
                         float* __restrict__ XC, int B, int Fh) {
  long long i = (long long)blockIdx.x * blockDim.x + threadIdx.x;
  const long long total = (long long)B * 2 * Fh;
  if (i >= total) return;
  const int r = (int)(i / (2 * Fh));
  const int c = (int)(i % (2 * Fh));
  XC[i] = (c < Fh) ? Xd[(size_t)r * Fh + c] : Xt[(size_t)r * Fh + (c - Fh)];
}

// ---------------------------------------------------------------------------
// Host side
// ---------------------------------------------------------------------------
static inline unsigned cdiv(long long a, long long b) { return (unsigned)((a + b - 1) / b); }

static void launch_gemm(const float* A, const float* W, const float* b, float* C,
                        int M, int K, int N, hipStream_t s) {
  dim3 grid(cdiv(N, BN), (unsigned)(M / BM));
  k_gemm_wmma<<<grid, 256, 0, s>>>(A, W, b, C, M, K, N);
}

static void run_gcn_branch(const float* X0, int Nn, int E,
                           const int* src, const int* dst, const int* batch,
                           const float* const Ws[3], const float* const bs[3],
                           const float* const gg[3], const float* const gb[3],
                           const float* const ga[3], const int Fd[4],
                           float* big0, float* big1, float* degbuf, float* normbuf,
                           float* pooled, hipStream_t s) {
  // degree -> dinv -> per-edge norm
  k_zero_f32<<<cdiv(Nn, TPB), TPB, 0, s>>>(degbuf, Nn);
  k_degree<<<cdiv(E, TPB), TPB, 0, s>>>(dst, degbuf, E);
  k_dinv<<<cdiv(Nn, TPB), TPB, 0, s>>>(degbuf, Nn);
  k_edge_norm<<<cdiv(E, TPB), TPB, 0, s>>>(src, dst, degbuf, normbuf, E);

  float* bufs[2] = {big0, big1};
  const float* X = X0;
  for (int i = 0; i < 3; ++i) {
    const int Fin = Fd[i], Fout = Fd[i + 1];
    float* H = bufs[i & 1];
    float* G = bufs[(i + 1) & 1];
    // H = X @ W            (X may alias G from previous layer: read before G is refilled)
    launch_gemm(X, Ws[i], nullptr, H, Nn, Fin, Fout, s);
    // G = bias; G[dst] += H[src] * norm
    const long long totN = (long long)Nn * Fout;
    const long long totE = (long long)E * Fout;
    k_fill_bias<<<cdiv(totN, TPB), TPB, 0, s>>>(G, bs[i], totN, Fout);
    k_scatter_add<<<cdiv(totE, TPB), TPB, 0, s>>>(H, src, dst, normbuf, G, totE, Fout);
    // out = relu(graphnorm(G)) -> reuse H buffer
    k_graphnorm_relu<<<1024, TPB, 0, s>>>(G, H, batch, Nn, Fout, gg[i], gb[i], ga[i]);
    X = H;
  }
  k_mean_pool<<<1024, TPB, 0, s>>>(X, pooled, batch, Nn, Fd[3]);
}

extern "C" void kernel_launch(void* const* d_in, const int* in_sizes, int n_in,
                              void* d_out, int out_size, void* d_ws, size_t ws_size,
                              hipStream_t stream) {
  (void)in_sizes; (void)n_in; (void)ws_size; (void)out_size;
  // Fixed problem sizes from setup_inputs()
  const int Nm = 49152, Np = 65536, Em = 196608, Ep = 524288, B = 1024;

  const float* x_mol  = (const float*)d_in[0];
  const float* x_prot = (const float*)d_in[1];
  auto F = [&](int i) { return (const float*)d_in[i]; };
  // params (make_params insertion order; _lin = {W,b}, _gn = {gamma,beta,alpha}, _bn = {gamma,beta})
  const float* dW[3] = {F(2), F(4), F(6)};   const float* db[3] = {F(3), F(5), F(7)};
  const float* tW[3] = {F(8), F(10), F(12)}; const float* tb[3] = {F(9), F(11), F(13)};
  const float* DGg[3] = {F(14), F(17), F(20)};
  const float* DGb[3] = {F(15), F(18), F(21)};
  const float* DGa[3] = {F(16), F(19), F(22)};
  const float* TGg[3] = {F(23), F(26), F(29)};
  const float* TGb[3] = {F(24), F(27), F(30)};
  const float* TGa[3] = {F(25), F(28), F(31)};
  const float *fcgd1W = F(32), *fcgd1b = F(33), *fcgd2W = F(34), *fcgd2b = F(35);
  const float *fcxt1W = F(36), *fcxt1b = F(37), *fcxt2W = F(38), *fcxt2b = F(39);
  const float *fc1W = F(40), *fc1b = F(41), *fc2W = F(42), *fc2b = F(43);
  const float *outW = F(44), *outb = F(45);
  const float *bn1g = F(46), *bn1b = F(47), *bn2g = F(48), *bn2b = F(49);
  const float *bn3g = F(50), *bn3b = F(51), *bn4g = F(52), *bn4b = F(53);
  const int* eim = (const int*)d_in[54];  // [2, Em]
  const int* batch_m = (const int*)d_in[55];
  const int* eip = (const int*)d_in[56];  // [2, Ep]
  const int* batch_p = (const int*)d_in[57];
  const int *sm = eim, *dm = eim + Em;
  const int *sp = eip, *dp = eip + Ep;

  // ---- workspace bump allocator ----
  char* wsp = (char*)d_ws;
  size_t off = 0;
  auto alloc = [&](size_t elems) -> float* {
    float* p = (float*)(wsp + off);
    off += ((elems * sizeof(float)) + 255) & ~(size_t)255;
    return p;
  };
  float* big0    = alloc((size_t)Np * 512);   // 134 MB, shared by both branches
  float* big1    = alloc((size_t)Np * 512);   // 134 MB
  float* degbuf  = alloc(Np);
  float* normbuf = alloc(Ep);
  float* pool_m  = alloc((size_t)B * 276);
  float* pool_p  = alloc((size_t)B * 512);
  float* fcA     = alloc((size_t)B * 1024);
  float* fcB     = alloc((size_t)B * 1024);
  float* xd      = alloc((size_t)B * 128);
  float* xt      = alloc((size_t)B * 128);
  float* xc      = alloc((size_t)B * 256);

  // ---- drug branch ----
  {
    const int Fd[4] = {69, 69, 138, 276};
    run_gcn_branch(x_mol, Nm, Em, sm, dm, batch_m, dW, db, DGg, DGb, DGa, Fd,
                   big0, big1, degbuf, normbuf, pool_m, stream);
    launch_gemm(pool_m, fcgd1W, fcgd1b, fcA, B, 276, 1024, stream);
    k_bn_relu<<<cdiv(1024, TPB), TPB, 0, stream>>>(fcA, fcB, bn1g, bn1b, B, 1024);
    launch_gemm(fcB, fcgd2W, fcgd2b, xd, B, 1024, 128, stream);
  }
  // ---- protein branch ----
  {
    const int Fd[4] = {33, 128, 256, 512};
    run_gcn_branch(x_prot, Np, Ep, sp, dp, batch_p, tW, tb, TGg, TGb, TGa, Fd,
                   big0, big1, degbuf, normbuf, pool_p, stream);
    launch_gemm(pool_p, fcxt1W, fcxt1b, fcA, B, 512, 1024, stream);
    k_bn_relu<<<cdiv(1024, TPB), TPB, 0, stream>>>(fcA, fcB, bn2g, bn2b, B, 1024);
    launch_gemm(fcB, fcxt2W, fcxt2b, xt, B, 1024, 128, stream);
  }
  // ---- head ----
  k_concat<<<cdiv((long long)B * 256, TPB), TPB, 0, stream>>>(xd, xt, xc, B, 128);
  launch_gemm(xc, fc1W, fc1b, fcA, B, 256, 1024, stream);
  k_bn_relu<<<cdiv(1024, TPB), TPB, 0, stream>>>(fcA, fcB, bn3g, bn3b, B, 1024);
  launch_gemm(fcB, fc2W, fc2b, fcA, B, 1024, 512, stream);
  k_bn_relu<<<cdiv(512, TPB), TPB, 0, stream>>>(fcA, fcB, bn4g, bn4b, B, 512);
  launch_gemm(fcB, outW, outb, (float*)d_out, B, 512, 1, stream);
}
